// EnhancedIPA2_27118423507359
// MI455X (gfx1250) — matrix operand order repaired and checked by reading, not compiled, not run on hardware
//
#include <hip/hip_runtime.h>
#include <hip/hip_bf16.h>

typedef __attribute__((ext_vector_type(16))) __bf16 v16bf;
typedef __attribute__((ext_vector_type(8)))  __bf16 v8bf;
typedef __attribute__((ext_vector_type(8)))  float  v8f;

#define BB 4
#define LL 1024
#define CSZ 384
#define HH 12
#define CC 16
#define PP 8
#define IPA_EPS 1e-12f

// ---------------------------------------------------------------------------
// Generic tiled GEMM: Y[M,N] = act(X[M,K] * W[K,N] + bias), bf16 WMMA, f32 acc.
// Block = 128 threads (4 waves). Block tile 64(M) x 64(N); wave w owns M-subtile w.
// Thread (warp,lane) owns one operand slot: 4x float4 loads for A, a batched
// 16-load column read for B (UNCONDITIONAL loads, index clamped; zeroing done
// by a VALU multiply so no per-element exec branches). Next K-chunk's loads
// are issued before the WMMAs so global latency hides under the matrix ops.
// M % 64 == 0, K % 32 == 0 required; N guarded.  act: 0 none, 1 ReLU, 2 sigmoid
// ---------------------------------------------------------------------------
__global__ __launch_bounds__(128)
void ipa_gemm_bias_act(const float* __restrict__ X, const float* __restrict__ W,
                       const float* __restrict__ bias, float* __restrict__ Y,
                       int M, int K, int N, int act)
{
    __shared__ __align__(32) __bf16 As[4][32][16];  // [m-sub][lane][elem] A order
    __shared__ __align__(32) __bf16 Bs[4][32][16];  // [n-sub][lane][elem] B order
    const int tid  = threadIdx.x;
    const int warp = tid >> 5;
    const int lane = tid & 31;
    const int m0 = blockIdx.y * 64;
    const int n0 = blockIdx.x * 64;

    // fill ownership: thread (warp, lane) fills As[warp][lane] and Bs[warp][lane]
    const int arow = m0 + warp * 16 + (lane & 15);
    const int aoff = (lane < 16) ? 0 : 8;    // A: lane<16 -> K {0..7,16..23}; +8 else
    const int gn   = n0 + warp * 16 + (lane & 15);
    const int boff = (lane < 16) ? 0 : 8;    // B: lane<16 -> K {0..7,16..23}; +8 else
    const int gnc  = (gn < N) ? gn : 0;      // clamped (always-valid) column
    const float msk = (gn < N) ? 1.0f : 0.0f;

    const float* xrow = X + (size_t)arow * K + aoff;

    float4 xa, xb, xc, xd;
    float wv[16];
    auto load_chunk = [&](int kb) {
        const float* xr = xrow + kb;
        xa = *(const float4*)(xr);
        xb = *(const float4*)(xr + 4);
        xc = *(const float4*)(xr + 16);
        xd = *(const float4*)(xr + 20);
        const float* wp = W + (size_t)(kb + boff) * N + gnc;
        #pragma unroll
        for (int e = 0; e < 16; ++e) wv[e] = wp[(size_t)((e < 8) ? e : e + 8) * N];
    };

    v8f acc[4] = {};
    load_chunk(0);

    for (int kb = 0; kb < K; kb += 32) {
        // ---- pack current chunk to registers, one 32B LDS store each ----
        v16bf av;
        av[0] = (__bf16)xa.x; av[1] = (__bf16)xa.y; av[2] = (__bf16)xa.z; av[3] = (__bf16)xa.w;
        av[4] = (__bf16)xb.x; av[5] = (__bf16)xb.y; av[6] = (__bf16)xb.z; av[7] = (__bf16)xb.w;
        av[8] = (__bf16)xc.x; av[9] = (__bf16)xc.y; av[10] = (__bf16)xc.z; av[11] = (__bf16)xc.w;
        av[12] = (__bf16)xd.x; av[13] = (__bf16)xd.y; av[14] = (__bf16)xd.z; av[15] = (__bf16)xd.w;
        v16bf bv;
        #pragma unroll
        for (int e = 0; e < 16; ++e) bv[e] = (__bf16)(wv[e] * msk);
        *(v16bf*)(&As[warp][lane][0]) = av;
        *(v16bf*)(&Bs[warp][lane][0]) = bv;

        __syncthreads();

        // ---- prefetch next chunk (overlaps the WMMAs below) ----
        if (kb + 32 < K) load_chunk(kb + 32);

        v16bf a = *(const v16bf*)(&As[warp][lane][0]);
        #pragma unroll
        for (int nt = 0; nt < 4; ++nt) {
            v16bf bm = *(const v16bf*)(&Bs[nt][lane][0]);
            acc[nt] = __builtin_amdgcn_wmma_f32_16x16x32_bf16(
                false, a, false, bm, (short)0, acc[nt], false, false);
        }
        __syncthreads();
    }

    // C/D layout: VGPR r, lane l -> M = r + 8*(l/16), N = l%16
    const int mhalf = (lane >> 4) << 3;
    #pragma unroll
    for (int nt = 0; nt < 4; ++nt) {
        int n = n0 + nt * 16 + (lane & 15);
        if (n < N) {
            float bvs = bias[n];
            #pragma unroll
            for (int r = 0; r < 8; ++r) {
                int m = m0 + warp * 16 + r + mhalf;
                float v = acc[nt][r] + bvs;
                if (act == 1) v = fmaxf(v, 0.0f);
                else if (act == 2) v = 1.0f / (1.0f + __expf(-v));
                Y[(size_t)m * N + n] = v;
            }
        }
    }
}

// ---------------------------------------------------------------------------
// Prep: frame transforms, gates, scale folding; emit WMMA-operand-ordered
// q-features (A layout), k-features (B layout), v-features (B layout),
// per-head mean coords (x,y,z,|.|^2) and v_pts mean.  One thread per (b,l,h).
// ---------------------------------------------------------------------------
__global__ __launch_bounds__(128)
void ipa_prep(const float* __restrict__ qproj, const float* __restrict__ kvproj,
              const float* __restrict__ qpproj, const float* __restrict__ kvpproj,
              const float* __restrict__ gproj,
              const float* __restrict__ rot, const float* __restrict__ trans,
              const float* __restrict__ head_weights, const float* __restrict__ geom_weight,
              __bf16* __restrict__ qfT, __bf16* __restrict__ kfB, __bf16* __restrict__ vfB,
              float* __restrict__ qc4, float* __restrict__ kc4, float* __restrict__ vmean)
{
    int t = blockIdx.x * blockDim.x + threadIdx.x;
    if (t >= BB * LL * HH) return;
    int h  = t % HH;
    int bl = t / HH;
    int l  = bl & (LL - 1);
    int b  = bl >> 10;

    float hw   = 1.0f / (1.0f + __expf(-head_weights[h]));
    float sc_s = hw * 0.25f;               // hw / sqrt(C=16)
    float sc_d = hw * geom_weight[1];      // hw * w_dir

    float R[9], T3[3];
    #pragma unroll
    for (int i = 0; i < 9; ++i) R[i] = rot[(size_t)bl * 9 + i];
    #pragma unroll
    for (int i = 0; i < 3; ++i) T3[i] = trans[(size_t)bl * 3 + i];

    float qfeat[64], kfeat[64], vfeat[64];
    #pragma unroll
    for (int i = 0; i < 64; ++i) { qfeat[i] = 0.f; kfeat[i] = 0.f; vfeat[i] = 0.f; }

    #pragma unroll
    for (int c = 0; c < CC; ++c) {
        qfeat[c] = qproj[(size_t)bl * (HH * CC) + h * CC + c] * sc_s;
        kfeat[c] = kvproj[(size_t)bl * (2 * HH * CC) + h * CC + c];
        vfeat[c] = kvproj[(size_t)bl * (2 * HH * CC) + HH * CC + h * CC + c];
    }

    float qcs0 = 0.f, qcs1 = 0.f, qcs2 = 0.f;
    float kcs0 = 0.f, kcs1 = 0.f, kcs2 = 0.f;
    float vms0 = 0.f, vms1 = 0.f, vms2 = 0.f;

    #pragma unroll
    for (int p = 0; p < PP; ++p) {
        const float* qp = qpproj + (size_t)bl * (6 * HH * PP) + (h * PP + p) * 6;
        float g = gproj[(size_t)bl * (HH * PP) + h * PP + p];
        float cx = (R[0]*qp[0] + R[1]*qp[1] + R[2]*qp[2] + T3[0]) * g;
        float cy = (R[3]*qp[0] + R[4]*qp[1] + R[5]*qp[2] + T3[1]) * g;
        float cz = (R[6]*qp[0] + R[7]*qp[1] + R[8]*qp[2] + T3[2]) * g;
        float dx = (R[0]*qp[3] + R[1]*qp[4] + R[2]*qp[5]) * g;
        float dy = (R[3]*qp[3] + R[4]*qp[4] + R[5]*qp[5]) * g;
        float dz = (R[6]*qp[3] + R[7]*qp[4] + R[8]*qp[5]) * g;
        qcs0 += cx; qcs1 += cy; qcs2 += cz;
        qfeat[16 + p * 3 + 0] = dx * sc_d;
        qfeat[16 + p * 3 + 1] = dy * sc_d;
        qfeat[16 + p * 3 + 2] = dz * sc_d;

        const float* kp = kvpproj + (size_t)bl * (12 * HH * PP) + (h * PP + p) * 6;
        float kcx = R[0]*kp[0] + R[1]*kp[1] + R[2]*kp[2] + T3[0];
        float kcy = R[3]*kp[0] + R[4]*kp[1] + R[5]*kp[2] + T3[1];
        float kcz = R[6]*kp[0] + R[7]*kp[1] + R[8]*kp[2] + T3[2];
        float kdx = R[0]*kp[3] + R[1]*kp[4] + R[2]*kp[5];
        float kdy = R[3]*kp[3] + R[4]*kp[4] + R[5]*kp[5];
        float kdz = R[6]*kp[3] + R[7]*kp[4] + R[8]*kp[5];
        kcs0 += kcx; kcs1 += kcy; kcs2 += kcz;
        kfeat[16 + p * 3 + 0] = kdx;
        kfeat[16 + p * 3 + 1] = kdy;
        kfeat[16 + p * 3 + 2] = kdz;

        const float* vp = kvpproj + (size_t)bl * (12 * HH * PP) + (6 * HH * PP) + (h * PP + p) * 6;
        float vcx = R[0]*vp[0] + R[1]*vp[1] + R[2]*vp[2] + T3[0];
        float vcy = R[3]*vp[0] + R[4]*vp[1] + R[5]*vp[2] + T3[1];
        float vcz = R[6]*vp[0] + R[7]*vp[1] + R[8]*vp[2] + T3[2];
        float vdx = R[0]*vp[3] + R[1]*vp[4] + R[2]*vp[5];
        float vdy = R[3]*vp[3] + R[4]*vp[4] + R[5]*vp[5];
        float vdz = R[6]*vp[3] + R[7]*vp[4] + R[8]*vp[5];
        vms0 += vcx; vms1 += vcy; vms2 += vcz;
        vfeat[16 + p * 6 + 0] = vcx;
        vfeat[16 + p * 6 + 1] = vcy;
        vfeat[16 + p * 6 + 2] = vcz;
        vfeat[16 + p * 6 + 3] = vdx;
        vfeat[16 + p * 6 + 4] = vdy;
        vfeat[16 + p * 6 + 5] = vdz;
    }

    const int bh = b * HH + h;
    {
        float x = qcs0 * 0.125f, y = qcs1 * 0.125f, z = qcs2 * 0.125f;
        float* o = qc4 + ((size_t)bh * LL + l) * 4;
        o[0] = x; o[1] = y; o[2] = z; o[3] = x*x + y*y + z*z;
        x = kcs0 * 0.125f; y = kcs1 * 0.125f; z = kcs2 * 0.125f;
        o = kc4 + ((size_t)bh * LL + l) * 4;
        o[0] = x; o[1] = y; o[2] = z; o[3] = x*x + y*y + z*z;
        float* vm = vmean + (size_t)t * 3;
        vm[0] = vms0 * 0.125f; vm[1] = vms1 * 0.125f; vm[2] = vms2 * 0.125f;
    }

    {
        int qt = l >> 4, m = l & 15;
        size_t base = ((size_t)(bh * 64 + qt) * 2) * 32 * 16;
        #pragma unroll
        for (int c2 = 0; c2 < 2; ++c2) {
            #pragma unroll
            for (int e = 0; e < 16; ++e) {
                int klo = c2 * 32 + ((e < 8) ? e : e + 8);
                qfT[base + (size_t)(c2 * 32 + m) * 16 + e]      = (__bf16)qfeat[klo];
                qfT[base + (size_t)(c2 * 32 + m + 16) * 16 + e] = (__bf16)qfeat[klo + 8];
                kfB[base + (size_t)(c2 * 32 + m) * 16 + e]      = (__bf16)kfeat[klo];
                kfB[base + (size_t)(c2 * 32 + m + 16) * 16 + e] = (__bf16)kfeat[klo + 8];
            }
        }
    }
    {
        int kch = l >> 5, kk = l & 31;
        int laneoff = (kk < 16) ? 0 : 16;
        int e2 = kk & 15;
        size_t base = ((size_t)(bh * 32 + kch) * 4) * 32 * 16;
        #pragma unroll
        for (int d = 0; d < 64; ++d) {
            int nt = d >> 4;
            int lanev = (d & 15) + laneoff;
            vfB[base + (size_t)(nt * 32 + lanev) * 16 + e2] = (__bf16)vfeat[d];
        }
    }
}

// ---------------------------------------------------------------------------
// Attention, flash-style: one wave per (b,h,16-query-row tile).
// Per 32-key block: two 16x16 score tiles (2 WMMAs each, K=64 feature dim),
// -w*sqrt(d2) positional term via VALU, online softmax (shfl_xor row
// reductions within 16-lane halves), accumulator rescale, bf16 prob tile
// staged in 1KB LDS in A-operand order, then 4 aggregation WMMAs.
// ---------------------------------------------------------------------------
__global__ __launch_bounds__(32)
void ipa_attention(const __bf16* __restrict__ qfT, const __bf16* __restrict__ kfB,
                   const __bf16* __restrict__ vfB,
                   const float* __restrict__ qc4, const float* __restrict__ kc4,
                   const float* __restrict__ head_weights,
                   const float* __restrict__ geom_weight,
                   float* __restrict__ O)
{
    __shared__ __align__(32) __bf16 probs[16][32];   // 1 KB, one wave per block
    const int lane = threadIdx.x;
    const int id = blockIdx.x;
    const int qt = id & 63;
    const int h  = (id >> 6) % HH;
    const int b  = id / (64 * HH);
    const int bh = b * HH + h;

    const float hw = 1.0f / (1.0f + __expf(-head_weights[h]));
    const float poscoef = hw * geom_weight[0];

    const __bf16* qbase = qfT + ((size_t)(bh * 64 + qt) * 2) * 32 * 16;
    v16bf a0 = *(const v16bf*)(qbase + (size_t)lane * 16);
    v16bf a1 = *(const v16bf*)(qbase + (size_t)(32 + lane) * 16);

    const int mhalf = (lane >> 4) << 3;
    const int n = lane & 15;
    const int m = lane & 15;
    const int off8 = (lane < 16) ? 0 : 16;

    float qcx[8], qcy[8], qcz[8], q2[8];
    #pragma unroll
    for (int r = 0; r < 8; ++r) {
        int qrow = qt * 16 + r + mhalf;
        const float* p = qc4 + ((size_t)bh * LL + qrow) * 4;
        qcx[r] = p[0]; qcy[r] = p[1]; qcz[r] = p[2]; q2[r] = p[3];
    }

    float mx[8], sum[8];
    #pragma unroll
    for (int r = 0; r < 8; ++r) { mx[r] = -3.0e38f; sum[r] = 0.0f; }
    v8f acc[4] = {};

    for (int kc = 0; kc < 32; ++kc) {
        // ---- two 16x16 score tiles ----
        float sc0[8], sc1[8];
        #pragma unroll
        for (int tt = 0; tt < 2; ++tt) {
            int kt = kc * 2 + tt;
            const __bf16* kb = kfB + ((size_t)(bh * 64 + kt) * 2) * 32 * 16;
            v16bf b0 = *(const v16bf*)(kb + (size_t)lane * 16);
            v16bf b1 = *(const v16bf*)(kb + (size_t)(32 + lane) * 16);
            v8f c = {};
            c = __builtin_amdgcn_wmma_f32_16x16x32_bf16(false, a0, false, b0, (short)0, c, false, false);
            c = __builtin_amdgcn_wmma_f32_16x16x32_bf16(false, a1, false, b1, (short)0, c, false, false);
            const float* kp = kc4 + ((size_t)bh * LL + kt * 16 + n) * 4;
            float kx = kp[0], ky = kp[1], kz = kp[2], k2 = kp[3];
            float* dst = tt ? sc1 : sc0;
            #pragma unroll
            for (int r = 0; r < 8; ++r) {
                float d2 = q2[r] + k2 - 2.0f * (qcx[r]*kx + qcy[r]*ky + qcz[r]*kz);
                d2 = fmaxf(d2, IPA_EPS);
                dst[r] = c[r] - poscoef * sqrtf(d2);
            }
        }
        // ---- online softmax update (rows live in 16-lane halves) ----
        #pragma unroll
        for (int r = 0; r < 8; ++r) {
            float v = fmaxf(sc0[r], sc1[r]);
            v = fmaxf(v, __shfl_xor(v, 1));
            v = fmaxf(v, __shfl_xor(v, 2));
            v = fmaxf(v, __shfl_xor(v, 4));
            v = fmaxf(v, __shfl_xor(v, 8));
            float nm = fmaxf(mx[r], v);
            float scale = __expf(mx[r] - nm);
            mx[r] = nm;
            float p0 = __expf(sc0[r] - nm);
            float p1 = __expf(sc1[r] - nm);
            float ps = p0 + p1;
            ps += __shfl_xor(ps, 1);
            ps += __shfl_xor(ps, 2);
            ps += __shfl_xor(ps, 4);
            ps += __shfl_xor(ps, 8);
            sum[r] = sum[r] * scale + ps;
            #pragma unroll
            for (int nt = 0; nt < 4; ++nt) acc[nt][r] *= scale;
            probs[r + mhalf][n]      = (__bf16)p0;
            probs[r + mhalf][16 + n] = (__bf16)p1;
        }
        // ---- aggregation: prob tile as A operand (per-wave DS is in-order) ----
        v8bf lo = *(const v8bf*)(&probs[m][off8]);
        v8bf hi = *(const v8bf*)(&probs[m][off8 + 8]);
        v16bf a = __builtin_shufflevector(lo, hi, 0, 1, 2, 3, 4, 5, 6, 7,
                                                  8, 9, 10, 11, 12, 13, 14, 15);
        const __bf16* vb = vfB + ((size_t)(bh * 32 + kc) * 4) * 32 * 16;
        #pragma unroll
        for (int nt = 0; nt < 4; ++nt) {
            v16bf bb = *(const v16bf*)(vb + (size_t)(nt * 32 + lane) * 16);
            acc[nt] = __builtin_amdgcn_wmma_f32_16x16x32_bf16(
                false, a, false, bb, (short)0, acc[nt], false, false);
        }
    }

    // ---- epilogue: fold 1/rowsum, store O[b,l,h,64] ----
    #pragma unroll
    for (int r = 0; r < 8; ++r) {
        float inv = 1.0f / sum[r];
        int ql = qt * 16 + r + mhalf;
        size_t base = (((size_t)b * LL + ql) * HH + h) * 64;
        #pragma unroll
        for (int nt = 0; nt < 4; ++nt) {
            O[base + nt * 16 + n] = acc[nt][r] * inv;
        }
    }
}

// ---------------------------------------------------------------------------
// Post: local coords, direction normalization, norms -> feats[4096, 864]
// ---------------------------------------------------------------------------
__global__ __launch_bounds__(128)
void ipa_post(const float* __restrict__ O, const float* __restrict__ vmean,
              float* __restrict__ feats)
{
    int t = blockIdx.x * blockDim.x + threadIdx.x;
    if (t >= BB * LL * HH) return;
    int h  = t % HH;
    int bl = t / HH;
    const float* o  = O + (size_t)t * 64;
    const float* vm = vmean + (size_t)t * 3;
    float* f = feats + (size_t)bl * (HH * (CC + 7 * PP));

    #pragma unroll
    for (int c = 0; c < CC; ++c) f[h * CC + c] = o[c];

    #pragma unroll
    for (int p = 0; p < PP; ++p) {
        const float* g = o + 16 + p * 6;
        float lx = g[0] - vm[0], ly = g[1] - vm[1], lz = g[2] - vm[2];
        float dn = sqrtf(g[3]*g[3] + g[4]*g[4] + g[5]*g[5]);
        float inv = 1.0f / fmaxf(dn, IPA_EPS);
        float lnorm = sqrtf(fmaxf(lx*lx + ly*ly + lz*lz, IPA_EPS));
        float* fp = f + HH * CC + (h * PP + p) * 7;
        fp[0] = lx; fp[1] = ly; fp[2] = lz;
        fp[3] = g[3] * inv; fp[4] = g[4] * inv; fp[5] = g[5] * inv;
        fp[6] = lnorm;
    }
}

// ---------------------------------------------------------------------------
extern "C" void kernel_launch(void* const* d_in, const int* in_sizes, int n_in,
                              void* d_out, int out_size, void* d_ws, size_t ws_size,
                              hipStream_t stream) {
    const float* s     = (const float*)d_in[0];
    const float* rot   = (const float*)d_in[1];
    const float* trans = (const float*)d_in[2];
    const float* Wq    = (const float*)d_in[3];
    const float* bq    = (const float*)d_in[4];
    const float* Wkv   = (const float*)d_in[5];
    const float* bkv   = (const float*)d_in[6];
    const float* Wqp   = (const float*)d_in[7];
    const float* bqp   = (const float*)d_in[8];
    const float* Wkvp  = (const float*)d_in[9];
    const float* bkvp  = (const float*)d_in[10];
    const float* Wg    = (const float*)d_in[11];
    const float* bg    = (const float*)d_in[12];
    const float* gw    = (const float*)d_in[13];
    const float* hws   = (const float*)d_in[14];
    const float* Wo    = (const float*)d_in[15];
    const float* bo    = (const float*)d_in[16];

    const int M = BB * LL;            // 4096
    char* ws = (char*)d_ws;
    size_t off = 0;
    auto alloc = [&](size_t bytes) -> char* {
        char* p = ws + off;
        off += (bytes + 255) & ~(size_t)255;
        return p;
    };
    float* qproj   = (float*)alloc((size_t)M * 192 * 4);
    float* kvproj  = (float*)alloc((size_t)M * 384 * 4);
    float* qpproj  = (float*)alloc((size_t)M * 576 * 4);
    float* kvpproj = (float*)alloc((size_t)M * 1152 * 4);
    float* gproj   = (float*)alloc((size_t)M * 96 * 4);
    float* qc4     = (float*)alloc((size_t)BB * HH * LL * 4 * 4);
    float* kc4     = (float*)alloc((size_t)BB * HH * LL * 4 * 4);
    float* vmean   = (float*)alloc((size_t)BB * LL * HH * 3 * 4);
    float* Obuf    = (float*)alloc((size_t)BB * LL * HH * 64 * 4);
    float* feats   = (float*)alloc((size_t)M * 864 * 4);
    __bf16* qfT    = (__bf16*)alloc((size_t)BB * HH * 64 * 2 * 32 * 16 * 2);
    __bf16* kfB    = (__bf16*)alloc((size_t)BB * HH * 64 * 2 * 32 * 16 * 2);
    __bf16* vfB    = (__bf16*)alloc((size_t)BB * HH * 32 * 4 * 32 * 16 * 2);

    dim3 blk(128);
    hipLaunchKernelGGL(ipa_gemm_bias_act, dim3(3, M / 64), blk, 0, stream,
                       s, Wq, bq, qproj, M, CSZ, 192, 0);
    hipLaunchKernelGGL(ipa_gemm_bias_act, dim3(6, M / 64), blk, 0, stream,
                       s, Wkv, bkv, kvproj, M, CSZ, 384, 0);
    hipLaunchKernelGGL(ipa_gemm_bias_act, dim3(9, M / 64), blk, 0, stream,
                       s, Wqp, bqp, qpproj, M, CSZ, 576, 1);
    hipLaunchKernelGGL(ipa_gemm_bias_act, dim3(18, M / 64), blk, 0, stream,
                       s, Wkvp, bkvp, kvpproj, M, CSZ, 1152, 1);
    hipLaunchKernelGGL(ipa_gemm_bias_act, dim3(2, M / 64), blk, 0, stream,
                       s, Wg, bg, gproj, M, CSZ, 96, 2);

    hipLaunchKernelGGL(ipa_prep, dim3((BB * LL * HH) / 128), blk, 0, stream,
                       qproj, kvproj, qpproj, kvpproj, gproj, rot, trans,
                       hws, gw, qfT, kfB, vfB, qc4, kc4, vmean);

    hipLaunchKernelGGL(ipa_attention, dim3(BB * HH * 64), dim3(32), 0, stream,
                       qfT, kfB, vfB, qc4, kc4, hws, gw, Obuf);

    hipLaunchKernelGGL(ipa_post, dim3((BB * LL * HH) / 128), blk, 0, stream,
                       Obuf, vmean, feats);

    hipLaunchKernelGGL(ipa_gemm_bias_act, dim3(6, M / 64), blk, 0, stream,
                       feats, Wo, bo, (float*)d_out, M, 864, CSZ, 0);
}
